// DeepInterestNetwork_74878459838783
// MI455X (gfx1250) — matrix operand order repaired and checked by reading, not compiled, not run on hardware
//
#include <hip/hip_runtime.h>
#include <hip/hip_bf16.h>
#include <stdint.h>

typedef __attribute__((ext_vector_type(16))) __bf16 v16bf;
typedef __attribute__((ext_vector_type(8)))  __bf16 v8bf;
typedef __attribute__((ext_vector_type(8)))  float  v8f;

#define B_   4096
#define T_   200
#define TP_  208     // T padded to 13 tiles of 16
#define E_   64
#define A1_  64
#define A2_  32
#define M1_  256
#define M2_  128

#define HS_  72      // bf16 row stride (K-pad) for hist/h1 in kernel A
#define WS_  72      // bf16 K-stride for transposed weights in kernel A
#define W1S_ 136     // kernel B: s_w1t K-stride (K=128 padded)
#define W2S_ 264     // kernel B: s_w2t K-stride (K=256 padded)
#define H1S_ 264     // kernel B: s_h1 row stride (256 padded)

static __device__ __forceinline__ __bf16 f2bf(float f) { return (__bf16)f; }

// Build a v16bf fragment from two contiguous 16-byte chunks.
static __device__ __forceinline__ v16bf ldfrag(const __bf16* p0, const __bf16* p1) {
    v8bf lo = *(const v8bf*)p0;
    v8bf hi = *(const v8bf*)p1;
    return __builtin_shufflevector(lo, hi, 0,1,2,3,4,5,6,7,8,9,10,11,12,13,14,15);
}

static __device__ __forceinline__ v8f wmma_bf16(v16bf a, v16bf b, v8f c) {
    return __builtin_amdgcn_wmma_f32_16x16x32_bf16(false, a, false, b, (short)0, c,
                                                   false, false);
}

// ---------------------------------------------------------------------------
// Kernel A: one workgroup (256 threads = 8 waves) per batch row.
// ---------------------------------------------------------------------------
__global__ void __launch_bounds__(256)
din_attention_kernel(const int* __restrict__ target,
                     const int* __restrict__ hist,
                     const unsigned char* __restrict__ mask,
                     const float* __restrict__ embed,
                     const float* __restrict__ aw1, const float* __restrict__ ab1,
                     const float* __restrict__ aw2, const float* __restrict__ ab2,
                     const float* __restrict__ aow, const float* __restrict__ aob,
                     __bf16* __restrict__ mlp_in)
{
    extern __shared__ char smem[];
    float*  s_hist   = (float*)smem;                  // [TP_][64]  f32
    float*  s_q      = s_hist + TP_ * E_;             // [64]
    float*  s_bias   = s_q + 64;                      // [64]
    float*  s_h2     = s_bias + 64;                   // [TP_][32]  f32
    float*  s_scores = s_h2 + TP_ * A2_;              // [TP_]
    float*  s_red    = s_scores + TP_;                // [256]
    float*  s_ab2    = s_red + 256;                   // [32]
    float*  s_aow    = s_ab2 + 32;                    // [32]
    __bf16* s_histb  = (__bf16*)(s_aow + 32);         // [TP_][HS_] bf16
    __bf16* s_Wt     = s_histb + TP_ * HS_;           // [64][WS_]  W_b^T
    __bf16* s_w2t    = s_Wt + 64 * WS_;               // [32][WS_]  aw2^T
    __bf16* s_h1     = s_w2t + 32 * WS_;              // [TP_][HS_] bf16

    const int b   = blockIdx.x;
    const int tid = threadIdx.x;

    // ---- query embedding ----
    const int tgt = target[b];
    if (tid < E_) s_q[tid] = embed[(long)tgt * E_ + tid];

    // ---- zero the f32 pad rows ----
    if (tid >= T_ && tid < TP_) {
        float4 z = {0.f, 0.f, 0.f, 0.f};
        float4* dst = (float4*)&s_hist[tid * E_];
        #pragma unroll
        for (int i = 0; i < E_ / 4; ++i) dst[i] = z;
    }

    // ---- async gather: 200 embedding rows -> LDS (ASYNCcnt-tracked DMA) ----
    for (int idx = tid; idx < T_ * 16; idx += 256) {
        const int row = idx >> 4, i = idx & 15;
        const int id  = hist[b * T_ + row];
        const unsigned lds = (unsigned)(uintptr_t)&s_hist[row * E_ + i * 4];
        const float* g = &embed[(long)id * E_ + i * 4];
        asm volatile("global_load_async_to_lds_b128 %0, %1, off"
                     :: "v"(lds), "v"(g) : "memory");
    }

    // ---- small weights (independent of the async data) ----
    if (tid < 32) { s_ab2[tid] = ab2[tid]; s_aow[tid] = aow[tid]; }
    for (int i = tid; i < A1_ * A2_; i += 256) {       // aw2^T staging
        const int k = i >> 5, j = i & 31;
        s_w2t[j * WS_ + k] = f2bf(aw2[i]);
    }

    asm volatile("s_wait_asynccnt 0" ::: "memory");
    __syncthreads();   // s_hist + s_q ready

    // ---- f32 hist -> packed bf16 copy (v8bf stores) ----
    for (int idx = tid; idx < TP_ * 8; idx += 256) {
        const int row = idx >> 3, c = idx & 7;
        const float4* src = (const float4*)&s_hist[row * E_ + c * 8];
        const float4 x0 = src[0], x1 = src[1];
        v8bf pk = { f2bf(x0.x), f2bf(x0.y), f2bf(x0.z), f2bf(x0.w),
                    f2bf(x1.x), f2bf(x1.y), f2bf(x1.z), f2bf(x1.w) };
        *(v8bf*)&s_histb[row * HS_ + c * 8] = pk;
    }

    // ---- fold aw1's 4 K-blocks into per-b W_b^T + bias_b ----
    for (int i = tid; i < 64 * 64; i += 256) {
        const int k = i >> 6, j = i & 63;
        const float w = aw1[k * A1_ + j] + aw1[(128 + k) * A1_ + j]
                      + s_q[k] * aw1[(192 + k) * A1_ + j];
        s_Wt[j * WS_ + k] = f2bf(w);
    }
    if (tid < 64) {
        float acc = ab1[tid];
        #pragma unroll 8
        for (int k = 0; k < 64; ++k)
            acc += s_q[k] * (aw1[(64 + k) * A1_ + tid] - aw1[(128 + k) * A1_ + tid]);
        s_bias[tid] = acc;
    }
    __syncthreads();

    const int wid  = tid >> 5;
    const int lane = tid & 31;
    const int m    = lane & 15;
    const int half = lane >> 4;

    // ---- GEMM1: h1[208,64] = relu(hist @ W_b + bias_b) ----
    for (int mt = wid; mt < 13; mt += 8) {
        const __bf16* arow = s_histb + (mt * 16 + m) * HS_;
        const v16bf a0 = ldfrag(arow + half * 8,      arow + half * 8 + 16);
        const v16bf a1 = ldfrag(arow + 32 + half * 8, arow + 32 + half * 8 + 16);
        #pragma unroll
        for (int nt = 0; nt < 4; ++nt) {
            const __bf16* bcol = s_Wt + (nt * 16 + m) * WS_;
            const v16bf b0 = ldfrag(bcol + half * 8,      bcol + half * 8 + 16);
            const v16bf b1 = ldfrag(bcol + 32 + half * 8, bcol + 32 + half * 8 + 16);
            v8f acc = {};
            acc = wmma_bf16(a0, b0, acc);
            acc = wmma_bf16(a1, b1, acc);
            const int n = nt * 16 + m;
            const float bj = s_bias[n];
            #pragma unroll
            for (int r = 0; r < 8; ++r) {
                float v = acc[r] + bj;
                v = v > 0.f ? v : 0.f;
                s_h1[(mt * 16 + half * 8 + r) * HS_ + n] = f2bf(v);
            }
        }
    }
    __syncthreads();

    // ---- GEMM2: h2[208,32] = relu(h1 @ aw2 + ab2) ----
    for (int mt = wid; mt < 13; mt += 8) {
        const __bf16* arow = s_h1 + (mt * 16 + m) * HS_;
        const v16bf a0 = ldfrag(arow + half * 8,      arow + half * 8 + 16);
        const v16bf a1 = ldfrag(arow + 32 + half * 8, arow + 32 + half * 8 + 16);
        #pragma unroll
        for (int nt = 0; nt < 2; ++nt) {
            const __bf16* bcol = s_w2t + (nt * 16 + m) * WS_;
            const v16bf b0 = ldfrag(bcol + half * 8,      bcol + half * 8 + 16);
            const v16bf b1 = ldfrag(bcol + 32 + half * 8, bcol + 32 + half * 8 + 16);
            v8f acc = {};
            acc = wmma_bf16(a0, b0, acc);
            acc = wmma_bf16(a1, b1, acc);
            const int n = nt * 16 + m;
            const float bj = s_ab2[n];
            #pragma unroll
            for (int r = 0; r < 8; ++r) {
                float v = acc[r] + bj;
                v = v > 0.f ? v : 0.f;
                s_h2[(mt * 16 + half * 8 + r) * A2_ + n] = v;
            }
        }
    }
    __syncthreads();

    // ---- score layer + mask ----
    if (tid < T_) {
        float s = aob[0];
        #pragma unroll
        for (int j = 0; j < A2_; ++j) s += s_h2[tid * A2_ + j] * s_aow[j];
        if (!mask[b * T_ + tid]) s = -1e9f;
        s_scores[tid] = s;
    }
    __syncthreads();

    // ---- softmax over T ----
    const float sv = (tid < T_) ? s_scores[tid] : -3.4e38f;
    s_red[tid] = sv;
    __syncthreads();
    for (int off = 128; off > 0; off >>= 1) {
        if (tid < off) s_red[tid] = fmaxf(s_red[tid], s_red[tid + off]);
        __syncthreads();
    }
    const float mx = s_red[0];
    __syncthreads();
    const float ev = (tid < T_) ? __expf(sv - mx) : 0.f;
    s_red[tid] = ev;
    __syncthreads();
    for (int off = 128; off > 0; off >>= 1) {
        if (tid < off) s_red[tid] += s_red[tid + off];
        __syncthreads();
    }
    const float inv = 1.f / s_red[0];
    __syncthreads();
    if (tid < T_) s_scores[tid] = ev * inv;
    __syncthreads();

    // ---- interest = sum_t w[t] * hist[t,:] ----
    const int c = tid & 63;
    const int p = tid >> 6;
    float acc = 0.f;
    for (int t = p; t < T_; t += 4) acc += s_scores[t] * s_hist[t * E_ + c];
    s_red[tid] = acc;
    __syncthreads();
    if (p == 0) {
        const float it = s_red[c] + s_red[c + 64] + s_red[c + 128] + s_red[c + 192];
        mlp_in[(long)b * 128 + c]      = f2bf(s_q[c]);
        mlp_in[(long)b * 128 + 64 + c] = f2bf(it);
    }
}

// ---------------------------------------------------------------------------
// Kernel B: final MLP 128 -> 256 -> 128 -> 1 over all B rows via WMMA.
// 128 threads (4 waves); each wave owns one 16-row M tile; block covers 64 rows.
// ---------------------------------------------------------------------------
__global__ void __launch_bounds__(128)
din_mlp_kernel(const __bf16* __restrict__ x,
               const float* __restrict__ mw1, const float* __restrict__ mb1,
               const float* __restrict__ mw2, const float* __restrict__ mb2,
               const float* __restrict__ ow,  const float* __restrict__ ob,
               float* __restrict__ out)
{
    extern __shared__ char smem[];
    __bf16* s_w1t = (__bf16*)smem;                 // [256][W1S_]  mw1^T
    __bf16* s_w2t = s_w1t + M1_ * W1S_;            // [128][W2S_]  mw2^T
    __bf16* s_h1  = s_w2t + M2_ * W2S_;            // [64][H1S_]
    float*  s_h2  = (float*)(s_h1 + 64 * H1S_);    // [64][128]
    float*  s_b1  = s_h2 + 64 * M2_;               // [256]
    float*  s_b2  = s_b1 + M1_;                    // [128]
    float*  s_ow  = s_b2 + M2_;                    // [128]

    const int tid = threadIdx.x;

    // stage transposed bf16 weights (coalesced global reads, scattered LDS)
    for (int i = tid; i < M2_ * M1_; i += 128) {   // mw1 [128][256]
        const int k = i >> 8, n = i & 255;
        s_w1t[n * W1S_ + k] = f2bf(mw1[i]);
    }
    for (int i = tid; i < M1_ * M2_; i += 128) {   // mw2 [256][128]
        const int k = i >> 7, n = i & 127;
        s_w2t[n * W2S_ + k] = f2bf(mw2[i]);
    }
    for (int i = tid; i < M1_; i += 128) s_b1[i] = mb1[i];
    if (tid < M2_) { s_b2[tid] = mb2[tid]; s_ow[tid] = ow[tid]; }
    __syncthreads();

    const int w    = tid >> 5;
    const int lane = tid & 31;
    const int m    = lane & 15;
    const int half = lane >> 4;
    const long r0  = (long)blockIdx.x * 64 + w * 16;

    // preload layer-1 A fragments straight from global (row-major bf16)
    const __bf16* xr = x + (r0 + m) * 128;
    v16bf afr[4];
    #pragma unroll
    for (int ks = 0; ks < 4; ++ks) {
        const int base = ks * 32 + half * 8;
        afr[ks] = ldfrag(xr + base, xr + base + 16);
    }

    // ---- layer 1: [16,128] @ [128,256], relu ----
    #pragma unroll
    for (int nt = 0; nt < 16; ++nt) {
        const __bf16* bc = s_w1t + (nt * 16 + m) * W1S_;
        v8f acc = {};
        #pragma unroll
        for (int ks = 0; ks < 4; ++ks) {
            const v16bf bf = ldfrag(bc + ks * 32 + half * 8,
                                    bc + ks * 32 + half * 8 + 16);
            acc = wmma_bf16(afr[ks], bf, acc);
        }
        const int n = nt * 16 + m;
        const float bj = s_b1[n];
        #pragma unroll
        for (int r = 0; r < 8; ++r) {
            float v = acc[r] + bj;
            v = v > 0.f ? v : 0.f;
            s_h1[(w * 16 + half * 8 + r) * H1S_ + n] = f2bf(v);
        }
    }
    __syncthreads();

    // preload layer-2 A fragments (K = 256 -> 8 fragments)
    const __bf16* hr = s_h1 + (w * 16 + m) * H1S_;
    v16bf a2[8];
    #pragma unroll
    for (int ks = 0; ks < 8; ++ks) {
        const int base = ks * 32 + half * 8;
        a2[ks] = ldfrag(hr + base, hr + base + 16);
    }

    // ---- layer 2: [16,256] @ [256,128], relu ----
    #pragma unroll
    for (int nt = 0; nt < 8; ++nt) {
        const __bf16* bc = s_w2t + (nt * 16 + m) * W2S_;
        v8f acc = {};
        #pragma unroll
        for (int ks = 0; ks < 8; ++ks) {
            const v16bf bf = ldfrag(bc + ks * 32 + half * 8,
                                    bc + ks * 32 + half * 8 + 16);
            acc = wmma_bf16(a2[ks], bf, acc);
        }
        const int n = nt * 16 + m;
        const float bj = s_b2[n];
        #pragma unroll
        for (int r = 0; r < 8; ++r) {
            float v = acc[r] + bj;
            v = v > 0.f ? v : 0.f;
            s_h2[(w * 16 + half * 8 + r) * M2_ + n] = v;
        }
    }
    __syncthreads();

    // ---- output: per-row 128-wide dot + bias ----
    if (lane < 16) {
        const int lrow = w * 16 + lane;
        float acc = ob[0];
        #pragma unroll 8
        for (int j = 0; j < M2_; ++j) acc += s_h2[lrow * M2_ + j] * s_ow[j];
        out[(long)blockIdx.x * 64 + lrow] = acc;
    }
}

// ---------------------------------------------------------------------------
extern "C" void kernel_launch(void* const* d_in, const int* in_sizes, int n_in,
                              void* d_out, int out_size, void* d_ws, size_t ws_size,
                              hipStream_t stream) {
    const int*           target = (const int*)d_in[0];
    const int*           hist   = (const int*)d_in[1];
    const unsigned char* mask   = (const unsigned char*)d_in[2];
    const float*         embed  = (const float*)d_in[3];
    const float* aw1 = (const float*)d_in[4];
    const float* ab1 = (const float*)d_in[5];
    const float* aw2 = (const float*)d_in[6];
    const float* ab2 = (const float*)d_in[7];
    const float* aow = (const float*)d_in[8];
    const float* aob = (const float*)d_in[9];
    const float* mw1 = (const float*)d_in[10];
    const float* mb1 = (const float*)d_in[11];
    const float* mw2 = (const float*)d_in[12];
    const float* mb2 = (const float*)d_in[13];
    const float* ow  = (const float*)d_in[14];
    const float* ob  = (const float*)d_in[15];

    __bf16* mlp_in = (__bf16*)d_ws;   // [B][128] bf16

    const size_t smemA =
        sizeof(float)  * (TP_ * E_ + 64 + 64 + TP_ * A2_ + TP_ + 256 + 32 + 32) +
        sizeof(__bf16) * (TP_ * HS_ + 64 * WS_ + 32 * WS_ + TP_ * HS_);
    const size_t smemB =
        sizeof(__bf16) * (M1_ * W1S_ + M2_ * W2S_ + 64 * H1S_) +
        sizeof(float)  * (64 * M2_ + M1_ + M2_ + M2_);

    din_attention_kernel<<<B_, 256, smemA, stream>>>(
        target, hist, mask, embed, aw1, ab1, aw2, ab2, aow, aob, mlp_in);

    din_mlp_kernel<<<B_ / 64, 128, smemB, stream>>>(
        mlp_in, mw1, mb1, mw2, mb2, ow, ob, (float*)d_out);
}
